// Gemma4VisionAttention_2035814498743
// MI455X (gfx1250) — matrix operand order repaired and checked by reading, not compiled, not run on hardware
//
#include <hip/hip_runtime.h>

#define SS   2048
#define HH   16
#define DDIM 72
#define HIDD 1152
#define DP   96          // padded head dim: 3 x 32 WMMA K-chunks
#define BSZ  4
#define MROWS (BSZ*SS)   // 8192

typedef __attribute__((ext_vector_type(16))) __bf16 v16bf;
typedef __attribute__((ext_vector_type(8)))  float  v8f;
typedef __attribute__((ext_vector_type(4)))  unsigned int v4u;
typedef __attribute__((ext_vector_type(8)))  unsigned int v8u;

__device__ __forceinline__ unsigned short f2bf(float f) {
  unsigned int u = __float_as_uint(f);
  return (unsigned short)((u + 0x7FFFu + ((u >> 16) & 1u)) >> 16);
}

__device__ __forceinline__ float waveSum32(float v) {
#pragma unroll
  for (int off = 16; off > 0; off >>= 1) v += __shfl_xor(v, off, 32);
  return v;
}

__device__ __forceinline__ v8f wmma_bf16(v16bf a, v16bf b, v8f c) {
  // (neg_a, A, neg_b, B, c_mod, C, reuse_a, reuse_b)
  return __builtin_amdgcn_wmma_f32_16x16x32_bf16(false, a, false, b, (short)0, c,
                                                 false, false);
}

union AFrag { v16bf v; unsigned int u[8]; uint4 q[2]; };

// ---- CDNA5 async global->LDS copy (ASYNCcnt path), 16B per lane ----
template <int OFF>
__device__ __forceinline__ void async_copy_b128(unsigned lds_off,
                                                unsigned long long gaddr) {
  asm volatile("global_load_async_to_lds_b128 %0, %1, off offset:%c2"
               :: "v"(lds_off), "v"(gaddr), "i"(OFF) : "memory");
}
__device__ __forceinline__ void async_wait0() {
  asm volatile("s_wait_asynccnt 0x0" ::: "memory");
}
__device__ __forceinline__ unsigned lds_off_of(const void* p) {
  // generic pointer to LDS = aperture(bits 63:32) | byte-offset(bits 31:0)
  return (unsigned)(size_t)p;
}

// ---- CDNA5 Tensor Data Mover: 2D tile global->LDS, one descriptor ----
// rows x cols (bf16), row stride = stride elements; wave-uniform arguments.
__device__ __forceinline__ void tdm_load_2d_bf16(unsigned lds_off,
                                                 unsigned long long gaddr,
                                                 unsigned cols, unsigned rows,
                                                 unsigned stride) {
  v4u g0; v8u g1;
  g0[0] = 1u;                                        // count=1, user mode
  g0[1] = lds_off;                                   // lds_addr (bytes)
  g0[2] = (unsigned)(gaddr & 0xFFFFFFFFull);         // global_addr[31:0]
  g0[3] = (unsigned)((gaddr >> 32) & 0x01FFFFFFull)  // global_addr[56:32]
          | (2u << 30);                              // type=2 ("image")
  g1[0] = 1u << 16;              // workgroup_mask=0, data_size=1 (2 bytes)
  g1[1] = (cols & 0xFFFFu) << 16;          // tensor_dim0[15:0]
  g1[2] = (cols >> 16) | ((rows & 0xFFFFu) << 16);   // td0 hi | tensor_dim1 lo
  g1[3] = (rows >> 16) | ((cols & 0xFFFFu) << 16);   // td1 hi | tile_dim0
  g1[4] = rows & 0xFFFFu;                  // tile_dim1 | tile_dim2=0
  g1[5] = stride;                          // tensor_dim0_stride[31:0]
  g1[6] = 0u;                              // stride hi | dim1_stride lo
  g1[7] = 0u;
  asm volatile("tensor_load_to_lds %0, %1" :: "s"(g0), "s"(g1) : "memory");
}

// ---------------- K0: fp32 -> bf16 convert ----------------
__global__ void cvt_f32_bf16(const float* __restrict__ s,
                             unsigned short* __restrict__ d, int n) {
  int i = blockIdx.x * blockDim.x + threadIdx.x;
  if (i < n) d[i] = f2bf(s[i]);
}

// -------- K1/K4: C[M,N](f32) = A[M,K](bf16) * Bw[N,K](bf16)^T --------
// block tile 128x128, 8 waves (4 along M x 2 along N), each wave 2x4 WMMA tiles
__global__ __launch_bounds__(256)
void gemm_bf16_nt(const unsigned short* __restrict__ A,
                  const unsigned short* __restrict__ Bw,
                  float* __restrict__ C, int M, int N, int K) {
  __shared__ __align__(16) unsigned short As[128 * 32]; // [m][k]
  __shared__ __align__(16) unsigned short Bs[32 * 128]; // [k][n] (transposed)

  const int tid = threadIdx.x;
  const int w = tid >> 5, l = tid & 31;
  const int wm = w & 3, wn = w >> 2;
  const int m0 = blockIdx.y * 128, n0 = blockIdx.x * 128;

  v8f acc[2][4] = {};

  for (int kk = 0; kk < K; kk += 32) {
    { // stage A tile via async global->LDS (straight copy, 32B per thread)
      int r = tid >> 1, c = (tid & 1) * 16;
      unsigned long long ga = (unsigned long long)&A[(size_t)(m0 + r) * K + kk + c];
      unsigned la = lds_off_of(&As[r * 32 + c]);
      async_copy_b128<0>(la, ga);
      async_copy_b128<16>(la, ga);
      if (kk + 32 < K) __builtin_prefetch(&A[(size_t)(m0 + r) * K + kk + 32 + c], 0, 1);
    }
    { // stage B tile transposed: Bs[k][n] = W[n0+n][kk+k]
      int n = tid & 127, kc = (tid >> 7) * 16;
      const uint4* g = (const uint4*)&Bw[(size_t)(n0 + n) * K + kk + kc];
      union { uint4 q[2]; unsigned short h[16]; } ub;
      ub.q[0] = g[0]; ub.q[1] = g[1];
#pragma unroll
      for (int i = 0; i < 16; ++i) Bs[(kc + i) * 128 + n] = ub.h[i];
      if (kk + 32 < K) __builtin_prefetch(&Bw[(size_t)(n0 + n) * K + kk + 32 + kc], 0, 1);
    }
    async_wait0();
    __syncthreads();

    AFrag af[2];
#pragma unroll
    for (int mt = 0; mt < 2; ++mt)
#pragma unroll
      for (int j = 0; j < 8; ++j) {
        int row = wm * 32 + mt * 16 + (l & 15);
        int col = ((j >> 2) << 4) + ((l >> 4) << 3) + ((j & 3) << 1);
        af[mt].u[j] = *(const unsigned int*)&As[row * 32 + col];
      }
    AFrag bf[4];
#pragma unroll
    for (int nt = 0; nt < 4; ++nt) {
#pragma unroll
      for (int j = 0; j < 8; ++j) {
        int k0 = ((j >> 2) << 4) + ((l >> 4) << 3) + ((j & 3) << 1);
        int n = wn * 64 + nt * 16 + (l & 15);
        bf[nt].u[j] = (unsigned int)Bs[k0 * 128 + n] | ((unsigned int)Bs[(k0 + 1) * 128 + n] << 16);
      }
    }
#pragma unroll
    for (int mt = 0; mt < 2; ++mt)
#pragma unroll
      for (int nt = 0; nt < 4; ++nt)
        acc[mt][nt] = wmma_bf16(af[mt].v, bf[nt].v, acc[mt][nt]);
    __syncthreads();
  }

#pragma unroll
  for (int mt = 0; mt < 2; ++mt)
#pragma unroll
    for (int nt = 0; nt < 4; ++nt)
#pragma unroll
      for (int r = 0; r < 8; ++r) {
        int row = m0 + wm * 32 + mt * 16 + r + ((l >> 4) << 3);
        int col = n0 + wn * 64 + nt * 16 + (l & 15);
        C[(size_t)row * N + col] = acc[mt][nt][r];
      }
}

// -------- K2: per-head RMS norm (fp32) + 2D RoPE + transpose/pad to bf16 -----
__global__ __launch_bounds__(256)
void qkv_post(const float* __restrict__ Qr, const float* __restrict__ Kr,
              const float* __restrict__ Vr, const float* __restrict__ Cos,
              const float* __restrict__ Sin, const float* __restrict__ Qw,
              const float* __restrict__ Kw,
              unsigned short* __restrict__ Qp, unsigned short* __restrict__ Kp,
              unsigned short* __restrict__ Vp) {
  const int bs = blockIdx.x;                 // b*S + s
  const int b = bs >> 11, s = bs & (SS - 1);
  const int w = threadIdx.x >> 5, l = threadIdx.x & 31;

  const size_t rowoff = (size_t)bs * HIDD;
  const float* cb = Cos + (size_t)bs * DDIM;
  const float* sb = Sin + (size_t)bs * DDIM;

#pragma unroll
  for (int hh = 0; hh < 2; ++hh) {
    const int h = w + hh * 8;
    const float* qb = Qr + rowoff + h * DDIM;
    const float* kb = Kr + rowoff + h * DDIM;
    const float* vb = Vr + rowoff + h * DDIM;

    float xq[3], xk[3], xv[3];
    int dd[3]; bool val[3];
    float aq = 0.f, ak = 0.f, av = 0.f;
#pragma unroll
    for (int i = 0; i < 3; ++i) {
      dd[i] = l + 32 * i; val[i] = dd[i] < DDIM;
      xq[i] = val[i] ? qb[dd[i]] : 0.f;
      xk[i] = val[i] ? kb[dd[i]] : 0.f;
      xv[i] = val[i] ? vb[dd[i]] : 0.f;
      aq += xq[i] * xq[i]; ak += xk[i] * xk[i]; av += xv[i] * xv[i];
    }
    const float eps = 1e-6f;
    float rq = rsqrtf(waveSum32(aq) * (1.f / DDIM) + eps);
    float rk = rsqrtf(waveSum32(ak) * (1.f / DDIM) + eps);
    float rv = rsqrtf(waveSum32(av) * (1.f / DDIM) + eps);

    const size_t ob = ((size_t)((b * HH + h) * SS) + s) * DP;
#pragma unroll
    for (int i = 0; i < 3; ++i) {
      if (!val[i]) continue;
      int d = dd[i], pd; float sgn;
      if (d < 18)      { pd = d + 18; sgn = -1.f; }
      else if (d < 36) { pd = d - 18; sgn =  1.f; }
      else if (d < 54) { pd = d + 18; sgn = -1.f; }
      else             { pd = d - 18; sgn =  1.f; }
      float c = cb[d], sn = sb[d];
      float yq  = xq[i]  * rq * Qw[d];
      float yqp = qb[pd] * rq * Qw[pd];
      Qp[ob + d] = f2bf(yq * c + sgn * yqp * sn);
      float yk  = xk[i]  * rk * Kw[d];
      float ykp = kb[pd] * rk * Kw[pd];
      Kp[ob + d] = f2bf(yk * c + sgn * ykp * sn);
      Vp[ob + d] = f2bf(xv[i] * rv);
    }
    if (l < DP - DDIM) {
      Qp[ob + DDIM + l] = 0; Kp[ob + DDIM + l] = 0; Vp[ob + DDIM + l] = 0;
    }
  }
}

// -------- K3: flash attention per (b,h, 128 q-rows); kv blocks of 64 --------
__global__ __launch_bounds__(256)
void attn_fused(const unsigned short* __restrict__ Qp,
                const unsigned short* __restrict__ Kp,
                const unsigned short* __restrict__ Vp,
                unsigned short* __restrict__ O) {
  __shared__ __align__(16) unsigned short Kt[DP * 64];     // [d][kv]
  __shared__ __align__(16) unsigned short Vt[64 * DP];     // [kv][d]
  __shared__ __align__(16) unsigned short Pl[8 * 16 * 64]; // per-wave P tile

  const int blk = blockIdx.x;
  const int qc = blk & 15;           // S/128 = 16 chunks
  const int bh = blk >> 4;           // b*H + h
  const int tid = threadIdx.x, w = tid >> 5, l = tid & 31;
  const int q0 = qc * 128;

  const unsigned short* Qb = Qp + (size_t)bh * SS * DP;
  const unsigned short* Kb = Kp + (size_t)bh * SS * DP;
  const unsigned short* Vb = Vp + (size_t)bh * SS * DP;

  // Q fragments (A-layout), 3 K-chunks of 32 covering DP=96, held in VGPRs
  AFrag qf[3];
  {
    const int qrow = q0 + w * 16 + (l & 15);
#pragma unroll
    for (int c = 0; c < 3; ++c)
#pragma unroll
      for (int j = 0; j < 8; ++j) {
        int col = c * 32 + ((j >> 2) << 4) + ((l >> 4) << 3) + ((j & 3) << 1);
        qf[c].u[j] = *(const unsigned int*)&Qb[(size_t)qrow * DP + col];
      }
  }

  float m_i[8], l_i[8];
  v8f oacc[6] = {};
#pragma unroll
  for (int r = 0; r < 8; ++r) { m_i[r] = -1e30f; l_i[r] = 0.f; }

  for (int kv0 = 0; kv0 < SS; kv0 += 64) {
    // V tile (64 rows x 96 bf16) via one TDM descriptor, issued by wave 0.
    if (tid < 32) {
      unsigned long long gv = (unsigned long long)&Vb[(size_t)kv0 * DP];
      tdm_load_2d_bf16(lds_off_of(&Vt[0]), gv, DP, 64, DP);
      __builtin_amdgcn_s_wait_tensorcnt(0);
    }
    { // K tile: manual transpose scatter Kt[d][kv]
      int r = tid >> 2, seg = tid & 3;
      const uint4* gk = (const uint4*)&Kb[(size_t)(kv0 + r) * DP + seg * 24];
      union { uint4 q[3]; unsigned short h[24]; } uk;
      uk.q[0] = gk[0]; uk.q[1] = gk[1]; uk.q[2] = gk[2];
#pragma unroll
      for (int i = 0; i < 24; ++i) Kt[(seg * 24 + i) * 64 + r] = uk.h[i];
    }
    __syncthreads();

    // scores: 16x64 per wave = 4 WMMA N-tiles, K-loop over 3 chunks of 32
    v8f sc[4];
#pragma unroll
    for (int nt = 0; nt < 4; ++nt) {
      v8f a = {};
#pragma unroll
      for (int c = 0; c < 3; ++c) {
        AFrag bu;
#pragma unroll
        for (int j = 0; j < 8; ++j) {
          int k0 = c * 32 + ((j >> 2) << 4) + ((l >> 4) << 3) + ((j & 3) << 1);
          int n = nt * 16 + (l & 15);
          bu.u[j] = (unsigned int)Kt[k0 * 64 + n] | ((unsigned int)Kt[(k0 + 1) * 64 + n] << 16);
        }
        a = wmma_bf16(qf[c].v, bu.v, a);
      }
      sc[nt] = a;
    }

    // online softmax: row = r + 8*(l>>4); xor-shuffles stay in 16-lane group
    float scl[8], rs[8];
#pragma unroll
    for (int r = 0; r < 8; ++r) {
      float mx = fmaxf(fmaxf(sc[0][r], sc[1][r]), fmaxf(sc[2][r], sc[3][r]));
#pragma unroll
      for (int off = 1; off < 16; off <<= 1) mx = fmaxf(mx, __shfl_xor(mx, off, 32));
      float mn = fmaxf(m_i[r], mx);
      scl[r] = __expf(m_i[r] - mn);
      m_i[r] = mn;
      rs[r] = 0.f;
    }

    unsigned short* Pw = &Pl[w * 1024];
#pragma unroll
    for (int nt = 0; nt < 4; ++nt)
#pragma unroll
      for (int r = 0; r < 8; ++r) {
        float p = __expf(sc[nt][r] - m_i[r]);
        rs[r] += p;
        Pw[(r + ((l >> 4) << 3)) * 64 + nt * 16 + (l & 15)] = f2bf(p);
      }
#pragma unroll
    for (int r = 0; r < 8; ++r) {
      float t = rs[r];
#pragma unroll
      for (int off = 1; off < 16; off <<= 1) t += __shfl_xor(t, off, 32);
      l_i[r] = l_i[r] * scl[r] + t;
#pragma unroll
      for (int dt = 0; dt < 6; ++dt) oacc[dt][r] *= scl[r];
    }
    __syncthreads();

    // re-load P in A-layout, then O += P * V (K-loop over 2 kv-chunks of 32)
    AFrag pf[2];
#pragma unroll
    for (int kc = 0; kc < 2; ++kc)
#pragma unroll
      for (int j = 0; j < 8; ++j) {
        int col = kc * 32 + ((j >> 2) << 4) + ((l >> 4) << 3) + ((j & 3) << 1);
        pf[kc].u[j] = *(const unsigned int*)&Pw[(l & 15) * 64 + col];
      }
#pragma unroll
    for (int dt = 0; dt < 6; ++dt)
#pragma unroll
      for (int kc = 0; kc < 2; ++kc) {
        AFrag bu;
#pragma unroll
        for (int j = 0; j < 8; ++j) {
          int k0 = kc * 32 + ((j >> 2) << 4) + ((l >> 4) << 3) + ((j & 3) << 1);
          int n = dt * 16 + (l & 15);
          bu.u[j] = (unsigned int)Vt[k0 * DP + n] | ((unsigned int)Vt[(k0 + 1) * DP + n] << 16);
        }
        oacc[dt] = wmma_bf16(pf[kc].v, bu.v, oacc[dt]);
      }
    __syncthreads();
  }

  // normalize and write [B,S,H*72] bf16 (drop the DP padding)
  const int b = bh >> 4, h = bh & 15;
#pragma unroll
  for (int dt = 0; dt < 6; ++dt) {
    int d = dt * 16 + (l & 15);
    if (d >= DDIM) continue;
#pragma unroll
    for (int r = 0; r < 8; ++r) {
      int srow = q0 + w * 16 + r + ((l >> 4) << 3);
      float o = oacc[dt][r] / l_i[r];
      O[((size_t)b * SS + srow) * HIDD + h * DDIM + d] = f2bf(o);
    }
  }
}

extern "C" void kernel_launch(void* const* d_in, const int* in_sizes, int n_in,
                              void* d_out, int out_size, void* d_ws, size_t ws_size,
                              hipStream_t stream) {
  (void)in_sizes; (void)n_in; (void)out_size; (void)ws_size;
  const float* hs = (const float*)d_in[0];
  const float* cs = (const float*)d_in[1];
  const float* sn = (const float*)d_in[2];
  const float* Wq = (const float*)d_in[3];
  const float* Wk = (const float*)d_in[4];
  const float* Wv = (const float*)d_in[5];
  const float* Wo = (const float*)d_in[6];
  const float* qw = (const float*)d_in[7];
  const float* kw = (const float*)d_in[8];
  float* out = (float*)d_out;

  char* ws = (char*)d_ws;
  const size_t SZ_XBF = (size_t)MROWS * HIDD * 2;        // bf16 activations
  const size_t SZ_WBF = (size_t)HIDD * HIDD * 2;         // bf16 weights
  const size_t SZ_RAW = (size_t)MROWS * HIDD * 4;        // fp32 raw QKV
  const size_t SZ_P   = (size_t)BSZ * HH * SS * DP * 2;  // padded bf16 QKV

  size_t o = 0;
  unsigned short* Xbf = (unsigned short*)(ws + o); o += SZ_XBF;
  unsigned short* Wqb = (unsigned short*)(ws + o); o += SZ_WBF;
  unsigned short* Wkb = (unsigned short*)(ws + o); o += SZ_WBF;
  unsigned short* Wvb = (unsigned short*)(ws + o); o += SZ_WBF;
  unsigned short* Wob = (unsigned short*)(ws + o); o += SZ_WBF;
  float* Qr = (float*)(ws + o); o += SZ_RAW;
  float* Kr = (float*)(ws + o); o += SZ_RAW;
  float* Vr = (float*)(ws + o); o += SZ_RAW;
  unsigned short* Qp = (unsigned short*)(ws + o); o += SZ_P;
  unsigned short* Kp = (unsigned short*)(ws + o); o += SZ_P;
  unsigned short* Vp = (unsigned short*)(ws + o); o += SZ_P;
  unsigned short* Obf = (unsigned short*)(ws + o); o += SZ_XBF;

  const int nX = MROWS * HIDD;
  const int nW = HIDD * HIDD;
  cvt_f32_bf16<<<(nX + 255) / 256, 256, 0, stream>>>(hs, Xbf, nX);
  cvt_f32_bf16<<<(nW + 255) / 256, 256, 0, stream>>>(Wq, Wqb, nW);
  cvt_f32_bf16<<<(nW + 255) / 256, 256, 0, stream>>>(Wk, Wkb, nW);
  cvt_f32_bf16<<<(nW + 255) / 256, 256, 0, stream>>>(Wv, Wvb, nW);
  cvt_f32_bf16<<<(nW + 255) / 256, 256, 0, stream>>>(Wo, Wob, nW);

  dim3 gg(HIDD / 128, MROWS / 128);  // (9, 64)
  gemm_bf16_nt<<<gg, 256, 0, stream>>>(Xbf, Wqb, Qr, MROWS, HIDD, HIDD);
  gemm_bf16_nt<<<gg, 256, 0, stream>>>(Xbf, Wkb, Kr, MROWS, HIDD, HIDD);
  gemm_bf16_nt<<<gg, 256, 0, stream>>>(Xbf, Wvb, Vr, MROWS, HIDD, HIDD);

  qkv_post<<<MROWS, 256, 0, stream>>>(Qr, Kr, Vr, cs, sn, qw, kw, Qp, Kp, Vp);

  attn_fused<<<BSZ * HH * (SS / 128), 256, 0, stream>>>(Qp, Kp, Vp, Obf);

  gemm_bf16_nt<<<gg, 256, 0, stream>>>(Obf, Wob, out, MROWS, HIDD, HIDD);
}
